// LongformerSiameseNetwork_57372173140239
// MI455X (gfx1250) — matrix operand order repaired and checked by reading, not compiled, not run on hardware
//
#include <hip/hip_runtime.h>

// ---------------------------------------------------------------------------
// CDNA5 (gfx1250) wave32 WMMA bf16 implementation of the Longformer-siamese
// forward pass.  All matmuls (QKV/O projections, FFN, attention scores and
// P*V) run on v_wmma_f32_16x16x32_bf16 with f32 accumulation.
// ---------------------------------------------------------------------------

typedef __bf16 bf16x16 __attribute__((ext_vector_type(16)));
typedef float  f32x8   __attribute__((ext_vector_type(8)));

union Frag16 { bf16x16 v; unsigned p[8]; };

__device__ __forceinline__ unsigned short f2bf(float f) {
  unsigned u = __float_as_uint(f);
  unsigned r = u + 0x7FFFu + ((u >> 16) & 1u);   // round-to-nearest-even
  return (unsigned short)(r >> 16);
}

__device__ __forceinline__ f32x8 v8zero() {
  f32x8 z;
#pragma unroll
  for (int i = 0; i < 8; ++i) z[i] = 0.0f;
  return z;
}

// A-matrix fragment (16x32 bf16): lane holds row (lane&15); VGPR p holds
// K = (p>=4)*16 + (lane>=16)*8 + 2*(p&3) .. +1   (ISA 7.12.2, 16-bit A)
// Source layout: [row][k] with row-stride `ld` (works for LDS or global).
__device__ __forceinline__ bf16x16 load_a_frag(const unsigned short* src,
                                               int row0, int k0, int ld) {
  const int lane = threadIdx.x & 31;
  const int half = lane >> 4;
  const unsigned short* rp = src + (size_t)(row0 + (lane & 15)) * ld + k0 + half * 8;
  Frag16 f;
#pragma unroll
  for (int p = 0; p < 8; ++p) {
    int kk = ((p >> 2) << 4) + ((p & 3) << 1);
    f.p[p] = *(const unsigned*)(rp + kk);
  }
  return f.v;
}

// B-matrix fragment (32x16 bf16): lane holds column (lane&15); VGPR p holds
// K = (p>=4)*16 + (lane>=16)*8 + 2*(p&3) .. +1   (matches A fragment K order)
// Source layout: column-major-as-rows, i.e. [n][k] with stride `ld`.
__device__ __forceinline__ bf16x16 load_b_frag(const unsigned short* src,
                                               int col0, int k0, int ld) {
  const int lane = threadIdx.x & 31;
  const int half = lane >> 4;
  const unsigned short* rp = src + (size_t)(col0 + (lane & 15)) * ld + k0 + half * 8;
  Frag16 f;
#pragma unroll
  for (int p = 0; p < 8; ++p) f.p[p] = *(const unsigned*)(rp + ((p >> 2) << 4) + ((p & 3) << 1));
  return f.v;
}

__device__ __forceinline__ f32x8 wmma_bf16(bf16x16 a, bf16x16 b, f32x8 c) {
  return __builtin_amdgcn_wmma_f32_16x16x32_bf16(false, a, false, b,
                                                 (short)0, c, false, false);
}

__device__ __forceinline__ float gelu_tanh(float x) {
  float x3 = x * x * x;
  float t = tanhf(0.7978845608028654f * (x + 0.044715f * x3));
  return 0.5f * x * (1.0f + t);
}

__device__ __forceinline__ float block_sum(float v, volatile float* sm) {
#pragma unroll
  for (int off = 16; off > 0; off >>= 1) v += __shfl_xor(v, off);
  __syncthreads();
  if ((threadIdx.x & 31) == 0) sm[threadIdx.x >> 5] = v;
  __syncthreads();
  float t = 0.0f;
#pragma unroll
  for (int i = 0; i < 8; ++i) t += sm[i];
  return t;
}

// ---------------------------------------------------------------------------
// GEMM: out[M,N] = A_bf16[M,K] * B_f32[K,N] + bias (+ residual) (GELU?).
// 128x128 tile / block, K-step 64 (2 WMMA k-steps per barrier pair),
// 8 waves in a 2x4 grid (64x32 per wave).
// B is converted f32->bf16 while being transposed into LDS [n][k].
// Next-stage global prefetch keeps HBM/L2 fetch overlapped with compute.
// ---------------------------------------------------------------------------
#define GLDK 72  // padded k-stride in LDS (shorts); 144B rows keep 16B alignment

template <bool GELU>
__global__ __launch_bounds__(256) void gemm_kernel(
    const unsigned short* __restrict__ A, const float* __restrict__ B,
    const float* __restrict__ bias, const float* __restrict__ resid,
    float* __restrict__ outF, unsigned short* __restrict__ outB,
    int M, int N, int K) {
  __shared__ unsigned short Asm[128 * GLDK];
  __shared__ unsigned short Bsm[128 * GLDK];  // transposed: [n][k]

  const int tid  = threadIdx.x;
  const int wave = tid >> 5;
  const int lane = tid & 31;
  const int half = lane >> 4;
  const int ncol = lane & 15;
  const int m0 = blockIdx.x * 128;
  const int n0 = blockIdx.y * 128;
  const int mw = (wave >> 2) * 64;
  const int nw = (wave & 3) * 32;

  f32x8 acc[4][2];
#pragma unroll
  for (int mi = 0; mi < 4; ++mi)
#pragma unroll
    for (int nj = 0; nj < 2; ++nj) acc[mi][nj] = v8zero();

  for (int k0 = 0; k0 < K; k0 += 64) {
    __syncthreads();
    // Stage A tile (128x64 bf16): 1024 chunks of 8 bf16.
#pragma unroll
    for (int c = 0; c < 4; ++c) {
      int idx = tid + c * 256;
      int r = idx >> 3, cc = (idx & 7) << 3;
      uint4 d = *(const uint4*)(A + (size_t)(m0 + r) * K + k0 + cc);
      *(uint4*)(&Asm[r * GLDK + cc]) = d;
    }
    // Stage B tile (64x128 f32 -> bf16, transposed into [n][k]).
#pragma unroll
    for (int c = 0; c < 8; ++c) {
      int idx = tid + c * 256;
      int r = idx >> 5, cc = (idx & 31) << 2;
      float4 d = *(const float4*)(B + (size_t)(k0 + r) * N + n0 + cc);
      Bsm[(cc + 0) * GLDK + r] = f2bf(d.x);
      Bsm[(cc + 1) * GLDK + r] = f2bf(d.y);
      Bsm[(cc + 2) * GLDK + r] = f2bf(d.z);
      Bsm[(cc + 3) * GLDK + r] = f2bf(d.w);
    }
    // Prefetch next K-stage (gfx1250 global_prefetch_b8).
    if (k0 + 64 < K) {
      __builtin_prefetch(A + (size_t)(m0 + (tid >> 1)) * K + k0 + 64 + ((tid & 1) << 5), 0, 1);
      __builtin_prefetch(B + (size_t)(k0 + 64 + (tid >> 3)) * N + n0 + ((tid & 7) << 4), 0, 1);
    }
    __syncthreads();

#pragma unroll
    for (int kk = 0; kk < 2; ++kk) {
      bf16x16 bf[2];
#pragma unroll
      for (int nj = 0; nj < 2; ++nj)
        bf[nj] = load_b_frag(Bsm, nw + nj * 16, kk * 32, GLDK);
#pragma unroll
      for (int mi = 0; mi < 4; ++mi) {
        bf16x16 af = load_a_frag(Asm, mw + mi * 16, kk * 32, GLDK);
#pragma unroll
        for (int nj = 0; nj < 2; ++nj)
          acc[mi][nj] = wmma_bf16(af, bf[nj], acc[mi][nj]);
      }
    }
  }

  // Epilogue: bias (+residual) (+GELU), write f32 and/or bf16.
#pragma unroll
  for (int mi = 0; mi < 4; ++mi) {
#pragma unroll
    for (int nj = 0; nj < 2; ++nj) {
      int n = n0 + nw + nj * 16 + ncol;
      float bval = bias[n];
#pragma unroll
      for (int r = 0; r < 8; ++r) {
        int m = m0 + mw + mi * 16 + r + half * 8;
        size_t off = (size_t)m * N + n;
        float v = acc[mi][nj][r] + bval;
        if (resid) v += resid[off];
        if (GELU) v = gelu_tanh(v);
        if (outF) outF[off] = v;
        if (outB) outB[off] = f2bf(v);
      }
    }
  }
}

// ---------------------------------------------------------------------------
// Sliding-window attention, flash style.  Block = (query block nb, head h).
// Wave w owns 32 query rows; loops over the 12 in-range 64-key sub-blocks.
// ---------------------------------------------------------------------------
#define ATP 72  // padded stride (shorts) for 64-wide tiles (144B, 16B-aligned)

__global__ __launch_bounds__(256) void sliding_attn_kernel(
    const unsigned short* __restrict__ Q, const unsigned short* __restrict__ Km,
    const unsigned short* __restrict__ V, const int* __restrict__ mask,
    unsigned short* __restrict__ O) {
  const int h  = blockIdx.y;
  const int nb = blockIdx.x;
  const int tid  = threadIdx.x;
  const int wave = tid >> 5;
  const int lane = tid & 31;
  const int half = lane >> 4;
  const int ncol = lane & 15;

  __shared__ unsigned short Klds[64 * ATP];           // [key][dh]  ([n][k] for K^T)
  __shared__ unsigned short Vlds[64 * ATP];           // [dh][key]  ([n][k] for V)
  __shared__ unsigned short Plds[8 * 32 * ATP];       // per-wave P buffer
  unsigned short* myP = Plds + wave * 32 * ATP;

  const int q0 = nb * 256 + wave * 32;

  // Q fragments straight from global: A layout, [row][k] with stride 768.
  bf16x16 aq[2][2];
#pragma unroll
  for (int mi = 0; mi < 2; ++mi)
#pragma unroll
    for (int kk = 0; kk < 2; ++kk)
      aq[mi][kk] = load_a_frag(Q + h * 64, q0 + mi * 16, kk * 32, 768);

  f32x8 oacc[2][4];
#pragma unroll
  for (int mi = 0; mi < 2; ++mi)
#pragma unroll
    for (int nd = 0; nd < 4; ++nd) oacc[mi][nd] = v8zero();
  float rm[2][8], rl[2][8];
#pragma unroll
  for (int mi = 0; mi < 2; ++mi)
#pragma unroll
    for (int r = 0; r < 8; ++r) { rm[mi][r] = -1e30f; rl[mi][r] = 0.0f; }

  for (int sub = 0; sub < 12; ++sub) {
    const int key0 = nb * 256 - 256 + sub * 64;
    if (key0 < 0 || key0 + 64 > 4096) continue;   // uniform per block
    __syncthreads();
    // Cooperative stage of K (row layout) and V (transposed) sub-tiles.
#pragma unroll
    for (int c = 0; c < 2; ++c) {
      int idx = tid + c * 256;
      int r = idx >> 3, cc = (idx & 7) << 3;
      uint4 kd = *(const uint4*)(Km + (size_t)(key0 + r) * 768 + h * 64 + cc);
      *(uint4*)(&Klds[r * ATP + cc]) = kd;
      uint4 vd = *(const uint4*)(V + (size_t)(key0 + r) * 768 + h * 64 + cc);
      const unsigned short* vs = (const unsigned short*)&vd;
#pragma unroll
      for (int e = 0; e < 8; ++e) Vlds[(cc + e) * ATP + r] = vs[e];
    }
    __syncthreads();

    // Scores S = Q * K^T  (32x64 per wave).
    f32x8 sfr[2][4];
#pragma unroll
    for (int mi = 0; mi < 2; ++mi)
#pragma unroll
      for (int nj = 0; nj < 4; ++nj) sfr[mi][nj] = v8zero();
#pragma unroll
    for (int kk = 0; kk < 2; ++kk) {
#pragma unroll
      for (int nj = 0; nj < 4; ++nj) {
        bf16x16 bk = load_b_frag(Klds, nj * 16, kk * 32, ATP);
#pragma unroll
        for (int mi = 0; mi < 2; ++mi)
          sfr[mi][nj] = wmma_bf16(aq[mi][kk], bk, sfr[mi][nj]);
      }
    }

    // Band mask (|key - q| <= 256), key-pad mask, 1/sqrt(64) scale.
    int kmv[4];
#pragma unroll
    for (int nj = 0; nj < 4; ++nj) kmv[nj] = mask[key0 + nj * 16 + ncol];
    const int dbase = (key0 + ncol) - (q0 + half * 8);  // d = dbase + 16nj - 16mi - r
#pragma unroll
    for (int mi = 0; mi < 2; ++mi) {
#pragma unroll
      for (int nj = 0; nj < 4; ++nj) {
        int dmn = dbase + nj * 16 - mi * 16;
#pragma unroll
        for (int r = 0; r < 8; ++r) {
          unsigned du = (unsigned)(dmn - r + 256);      // in-band iff du <= 512
          bool ok = (du <= 512u) && (kmv[nj] != 0);
          sfr[mi][nj][r] = ok ? sfr[mi][nj][r] * 0.125f : -1e30f;
        }
      }
    }

    // Online softmax update + P -> LDS (bf16, A layout source).
#pragma unroll
    for (int mi = 0; mi < 2; ++mi) {
#pragma unroll
      for (int r = 0; r < 8; ++r) {
        float vmx = -1e30f;
#pragma unroll
        for (int nj = 0; nj < 4; ++nj) vmx = fmaxf(vmx, sfr[mi][nj][r]);
#pragma unroll
        for (int off = 1; off < 16; off <<= 1) vmx = fmaxf(vmx, __shfl_xor(vmx, off));
        float mnew  = fmaxf(rm[mi][r], vmx);
        float alpha = __expf(rm[mi][r] - mnew);
        rm[mi][r] = mnew;
        rl[mi][r] *= alpha;
#pragma unroll
        for (int nd = 0; nd < 4; ++nd) oacc[mi][nd][r] *= alpha;
        float ps = 0.0f;
#pragma unroll
        for (int nj = 0; nj < 4; ++nj) {
          float sv = sfr[mi][nj][r];
          float pv = __expf(sv - mnew);
          if (sv < -1e29f) pv = 0.0f;   // fully-masked guard
          ps += pv;
          myP[(mi * 16 + r + half * 8) * ATP + nj * 16 + ncol] = f2bf(pv);
        }
#pragma unroll
        for (int off = 1; off < 16; off <<= 1) ps += __shfl_xor(ps, off);
        rl[mi][r] += ps;
      }
    }

    // O += P * V  (same-wave DS ops are in order: safe to reload myP).
#pragma unroll
    for (int kk = 0; kk < 2; ++kk) {
      bf16x16 ap[2];
#pragma unroll
      for (int mi = 0; mi < 2; ++mi) ap[mi] = load_a_frag(myP, mi * 16, kk * 32, ATP);
#pragma unroll
      for (int nd = 0; nd < 4; ++nd) {
        bf16x16 bv = load_b_frag(Vlds, nd * 16, kk * 32, ATP);
#pragma unroll
        for (int mi = 0; mi < 2; ++mi)
          oacc[mi][nd] = wmma_bf16(ap[mi], bv, oacc[mi][nd]);
      }
    }
  }

  // Normalize and write context in (S, H*Dh) layout.
#pragma unroll
  for (int mi = 0; mi < 2; ++mi) {
#pragma unroll
    for (int nd = 0; nd < 4; ++nd) {
#pragma unroll
      for (int r = 0; r < 8; ++r) {
        int qg = q0 + mi * 16 + r + half * 8;
        float denom = rl[mi][r];
        float v = (denom > 0.0f) ? oacc[mi][nd][r] / denom : 0.0f;
        O[(size_t)qg * 768 + h * 64 + nd * 16 + ncol] = f2bf(v);
      }
    }
  }
}

// ---------------------------------------------------------------------------
// Embedding + LayerNorm, LayerNorm, mean-pool, sigmoid head.
// ---------------------------------------------------------------------------
__device__ __forceinline__ void ln_core(float v[3], const float* g, const float* b,
                                        float* outF, unsigned short* outB,
                                        size_t rowbase, float* sm) {
  float S = block_sum(v[0] + v[1] + v[2], sm);
  float mean = S * (1.0f / 768.0f);
  float d0 = v[0] - mean, d1 = v[1] - mean, d2 = v[2] - mean;
  float S2 = block_sum(d0 * d0 + d1 * d1 + d2 * d2, sm);
  float inv = rsqrtf(S2 * (1.0f / 768.0f) + 1e-5f);
#pragma unroll
  for (int j = 0; j < 3; ++j) {
    int i = threadIdx.x + j * 256;
    float y = (v[j] - mean) * inv * g[i] + b[i];
    if (outF) outF[rowbase + i] = y;
    if (outB) outB[rowbase + i] = f2bf(y);
  }
}

__global__ __launch_bounds__(256) void embed_ln_kernel(
    const int* __restrict__ ids, const float* __restrict__ wemb,
    const float* __restrict__ pemb, const float* __restrict__ g,
    const float* __restrict__ b, float* __restrict__ outF,
    unsigned short* __restrict__ outB) {
  __shared__ float sm[8];
  const int row = blockIdx.x;
  const size_t id = (size_t)ids[row];
  float v[3];
#pragma unroll
  for (int j = 0; j < 3; ++j) {
    int i = threadIdx.x + j * 256;
    v[j] = wemb[id * 768 + i] + pemb[(size_t)row * 768 + i];
  }
  ln_core(v, g, b, outF, outB, (size_t)row * 768, sm);
}

__global__ __launch_bounds__(256) void ln_kernel(
    const float* __restrict__ in, const float* __restrict__ g,
    const float* __restrict__ b, float* __restrict__ outF,
    unsigned short* __restrict__ outB) {
  __shared__ float sm[8];
  const int row = blockIdx.x;
  float v[3];
#pragma unroll
  for (int j = 0; j < 3; ++j) v[j] = in[(size_t)row * 768 + threadIdx.x + j * 256];
  ln_core(v, g, b, outF, outB, (size_t)row * 768, sm);
}

__global__ __launch_bounds__(256) void mean_pool_kernel(const float* __restrict__ x,
                                                        float* __restrict__ out) {
  int col = blockIdx.x * 256 + threadIdx.x;
  if (col >= 768) return;
  float s = 0.0f;
  for (int r = 0; r < 4096; ++r) s += x[(size_t)r * 768 + col];
  out[col] = s * (1.0f / 4096.0f);
}

__global__ __launch_bounds__(256) void final_head_kernel(
    const float* __restrict__ p1, const float* __restrict__ p2,
    const float* __restrict__ w, const float* __restrict__ b,
    float* __restrict__ out) {
  __shared__ float sm[8];
  float s = 0.0f;
  for (int i = threadIdx.x; i < 768; i += 256) s += (p1[i] - p2[i]) * w[i];
  s = block_sum(s, sm);
  if (threadIdx.x == 0) out[0] = 1.0f / (1.0f + __expf(-(s + b[0])));
}

// ---------------------------------------------------------------------------
// Orchestration.
// ---------------------------------------------------------------------------
extern "C" void kernel_launch(void* const* d_in, const int* in_sizes, int n_in,
                              void* d_out, int out_size, void* d_ws, size_t ws_size,
                              hipStream_t stream) {
  (void)in_sizes; (void)n_in; (void)out_size; (void)ws_size;
  const int S = 4096, DM = 768, FF = 3072, L = 4;

  const int* ids[2]   = {(const int*)d_in[0], (const int*)d_in[1]};
  const int* amask[2] = {(const int*)d_in[2], (const int*)d_in[3]};
  const float* wemb = (const float*)d_in[4];
  const float* pemb = (const float*)d_in[5];
  const float* embg = (const float*)d_in[6];
  const float* embb = (const float*)d_in[7];
  const float* wq = (const float*)d_in[8];
  const float* bq = (const float*)d_in[9];
  const float* wk = (const float*)d_in[10];
  const float* bk = (const float*)d_in[11];
  const float* wv = (const float*)d_in[12];
  const float* bv = (const float*)d_in[13];
  const float* wo = (const float*)d_in[14];
  const float* bo = (const float*)d_in[15];
  const float* ln1g = (const float*)d_in[16];
  const float* ln1b = (const float*)d_in[17];
  const float* w1 = (const float*)d_in[18];
  const float* b1 = (const float*)d_in[19];
  const float* w2 = (const float*)d_in[20];
  const float* b2 = (const float*)d_in[21];
  const float* ln2g = (const float*)d_in[22];
  const float* ln2b = (const float*)d_in[23];
  const float* dw = (const float*)d_in[24];
  const float* db = (const float*)d_in[25];

  char* wp = (char*)d_ws;
  auto carve = [&](size_t bytes) -> char* {
    char* r = wp;
    wp += (bytes + 255) & ~(size_t)255;
    return r;
  };
  float*          xF   = (float*)carve((size_t)S * DM * 4);
  unsigned short* xB   = (unsigned short*)carve((size_t)S * DM * 2);
  float*          x2F  = (float*)carve((size_t)S * DM * 4);
  unsigned short* x2B  = (unsigned short*)carve((size_t)S * DM * 2);
  float*          tmpF = (float*)carve((size_t)S * DM * 4);
  unsigned short* qB   = (unsigned short*)carve((size_t)S * DM * 2);
  unsigned short* kB   = (unsigned short*)carve((size_t)S * DM * 2);
  unsigned short* vB   = (unsigned short*)carve((size_t)S * DM * 2);
  unsigned short* ctxB = (unsigned short*)carve((size_t)S * DM * 2);
  unsigned short* hB   = (unsigned short*)carve((size_t)S * FF * 2);
  float*          pool = (float*)carve((size_t)2 * DM * 4);

  dim3 blk(256);
  dim3 gP(S / 128, DM / 128);   // 4096x768 GEMMs
  dim3 gF1(S / 128, FF / 128);  // 4096x3072 GEMM
  dim3 gAt(16, 12);             // (query blocks, heads)

  for (int s = 0; s < 2; ++s) {
    embed_ln_kernel<<<S, blk, 0, stream>>>(ids[s], wemb, pemb, embg, embb, xF, xB);
    for (int l = 0; l < L; ++l) {
      const size_t oDD = (size_t)l * DM * DM;
      const size_t oDF = (size_t)l * DM * FF;
      gemm_kernel<false><<<gP, blk, 0, stream>>>(xB, wq + oDD, bq + l * DM,
                                                 nullptr, nullptr, qB, S, DM, DM);
      gemm_kernel<false><<<gP, blk, 0, stream>>>(xB, wk + oDD, bk + l * DM,
                                                 nullptr, nullptr, kB, S, DM, DM);
      gemm_kernel<false><<<gP, blk, 0, stream>>>(xB, wv + oDD, bv + l * DM,
                                                 nullptr, nullptr, vB, S, DM, DM);
      sliding_attn_kernel<<<gAt, blk, 0, stream>>>(qB, kB, vB, amask[s], ctxB);
      gemm_kernel<false><<<gP, blk, 0, stream>>>(ctxB, wo + oDD, bo + l * DM,
                                                 xF, tmpF, nullptr, S, DM, DM);
      ln_kernel<<<S, blk, 0, stream>>>(tmpF, ln1g + l * DM, ln1b + l * DM, x2F, x2B);
      gemm_kernel<true><<<gF1, blk, 0, stream>>>(x2B, w1 + oDF, b1 + l * FF,
                                                 nullptr, nullptr, hB, S, FF, DM);
      gemm_kernel<false><<<gP, blk, 0, stream>>>(hB, w2 + oDF, b2 + l * DM,
                                                 x2F, tmpF, nullptr, S, DM, FF);
      ln_kernel<<<S, blk, 0, stream>>>(tmpF, ln2g + l * DM, ln2b + l * DM, xF, xB);
    }
    mean_pool_kernel<<<3, blk, 0, stream>>>(xF, pool + s * DM);
  }
  final_head_kernel<<<1, blk, 0, stream>>>(pool, pool + DM, dw, db, (float*)d_out);
}
